// SelectiveAttn_88235808129267
// MI455X (gfx1250) — matrix operand order repaired and checked by reading, not compiled, not run on hardware
//
#include <hip/hip_runtime.h>

// Selective top-k attention decode for gfx1250 (MI455X).
// bf16 WMMA flash attention; K gathered via global_load_async_to_lds_b128 (double-buffered),
// V gathered straight from global (B-fragment layout is 64B-coalesced per half-wave).
// Grid: (T=64, NUM_KV_HEADS=4). Block: 128 threads = 4 wave32 waves.
// Each wave owns 4 select-blocks (256 KV positions), processed in 32-position chunks.

#define TQ      64
#define NKV     4
#define GRP     16      // q heads per kv head
#define DQK     128
#define DV      128
#define SELSZ   64
#define TOPK    16
#define MAXCTX  4096
#define SCALE   0.08838834764831845f  // 1/sqrt(128)
#define ROWF    132                   // staged K row stride in floats (128 + 4 pad)

typedef __attribute__((ext_vector_type(16))) __bf16 v16bf;
typedef __attribute__((ext_vector_type(8)))  float  v8f;

// Pointer types matching the async-to-LDS builtin prototype:
//   void __builtin_amdgcn_global_load_async_to_lds_b128(AS1 v4i*, AS3 v4i*, int, int)
typedef int v4i __attribute__((vector_size(16)));
typedef __attribute__((address_space(1))) v4i* gbl_v4i_p;
typedef __attribute__((address_space(3))) v4i* lds_v4i_p;

#if __has_builtin(__builtin_amdgcn_global_load_async_to_lds_b128)
#define HAVE_ASYNC 1
#else
#define HAVE_ASYNC 0
#endif

__device__ __forceinline__ void stage16(const void* g, void* l) {
#if HAVE_ASYNC
  __builtin_amdgcn_global_load_async_to_lds_b128((gbl_v4i_p)g, (lds_v4i_p)l, 0, 0);
#else
  *(float4*)l = *(const float4*)g;  // synchronous fallback; compiler orders via dscnt
#endif
}

// Allow the newest chunk's 32 K-row loads to remain outstanding; anything older is done
// (async loads complete in order).
__device__ __forceinline__ void wait_stage_31() {
#if HAVE_ASYNC
#if __has_builtin(__builtin_amdgcn_s_wait_asynccnt)
  __builtin_amdgcn_s_wait_asynccnt(31);
#else
  asm volatile("s_wait_asynccnt 31" ::: "memory");
#endif
#endif
  asm volatile("" ::: "memory");
}

__device__ __forceinline__ void wait_stage_0() {
#if HAVE_ASYNC
#if __has_builtin(__builtin_amdgcn_s_wait_asynccnt)
  __builtin_amdgcn_s_wait_asynccnt(0);
#else
  asm volatile("s_wait_asynccnt 0" ::: "memory");
#endif
#endif
  asm volatile("" ::: "memory");
}

__global__ __launch_bounds__(128)
void SelAttn_gfx1250_kernel(const float* __restrict__ q,
                            const int*   __restrict__ sel_idx,
                            const float* __restrict__ k_cache,
                            const float* __restrict__ v_cache,
                            const int*   __restrict__ req_to_token,
                            const int*   __restrict__ seq_lens,
                            float*       __restrict__ out)
{
  // K staging: 4 waves x double-buffer x 32 rows x ROWF floats  (~135 KB)
  __shared__ float kbuf[4 * 2 * 32 * ROWF];
  __shared__ float o_ws[4 * GRP * DV];   // per-wave partial O for flash combine
  __shared__ float sm_s[4 * GRP];        // per-wave running max
  __shared__ float sl_s[4 * GRP];        // per-wave running sum

  const int t    = blockIdx.x;
  const int h    = blockIdx.y;
  const int tid  = threadIdx.x;
  const int w    = tid >> 5;
  const int lane = tid & 31;
  const int lo   = lane & 15;
  const bool hi  = lane >= 16;
  const int hi8  = hi ? 8 : 0;

  const int  seqlen = seq_lens[t];
  const int* rt = req_to_token + (size_t)t * MAXCTX;
  const int* si = sel_idx + ((size_t)t * NKV + h) * TOPK;

  // ---- Q fragments (B operand of S^T = K * Q^T), built once from global ----
  // B 32x16 bf16 layout: lane<16 -> col N=lane, K={0..7,16..23}; lane>=16 -> col N=lane-16, K={8..15,24..31}
  v16bf qf[4];
  {
    const float* qbase = q + (size_t)t * (NKV * GRP * DQK) + ((size_t)h * GRP + lo) * DQK;
    #pragma unroll
    for (int kk = 0; kk < 4; ++kk) {
      const float* qp = qbase + kk * 32 + (hi ? 8 : 0);
      v16bf b;
      #pragma unroll
      for (int j = 0; j < 16; ++j) b[j] = (__bf16)qp[j + (j & 8)];
      qf[kk] = b;
    }
  }

  v8f ot[8];
  #pragma unroll
  for (int d = 0; d < 8; ++d) {
    #pragma unroll
    for (int r = 0; r < 8; ++r) ot[d][r] = 0.0f;
  }
  float m_run = -3.0e38f, l_run = 0.0f;

  float* kw = kbuf + w * (2 * 32 * ROWF);

  // ---- stage one 32-position chunk of gathered K into LDS (async, 32 issues) ----
  auto stage_chunk = [&](int cc) {
    const int blk     = 4 * w + (cc >> 1);
    const int absbase = si[blk] * SELSZ + (cc & 1) * 32;
    float* kb = kw + (cc & 1) * (32 * ROWF);
    #pragma unroll 4
    for (int row = 0; row < 32; ++row) {
      int pos  = absbase + row;
      int posc = pos < (MAXCTX - 1) ? pos : (MAXCTX - 1);
      int tok  = rt[posc];
      const float* gk = k_cache + ((size_t)tok * NKV + h) * DQK;
      // 32 lanes x 16B = full 512B row, perfectly coalesced
      stage16((const char*)gk + lane * 16, (char*)(kb + row * ROWF) + lane * 16);
    }
  };

  // ---- flash-attention step over one 32-position chunk ----
  auto process_chunk = [&](int cc) {
    const int blk     = 4 * w + (cc >> 1);
    const int absbase = si[blk] * SELSZ + (cc & 1) * 32;
    const float* kb = kw + (cc & 1) * (32 * ROWF);

    // Per-lane V row offsets (in floats) for this chunk's B fragments:
    // lane<16 serves K(pos) {0..7,16..23}, lane>=16 serves pos {8..15,24..31}.
    const int pos0 = hi ? 8 : 0;
    unsigned voffs[16];
    #pragma unroll
    for (int j = 0; j < 16; ++j) {
      int pos  = absbase + pos0 + j + (j & 8);
      int posc = pos < (MAXCTX - 1) ? pos : (MAXCTX - 1);
      int tok  = rt[posc];
      voffs[j] = ((unsigned)tok * NKV + h) * DV;
    }

    // S^T tiles: D[M=pos][N=head] = K_tile * Q^T, two 16-pos tiles, 4 bf16 WMMAs each
    v8f c0, c1;
    #pragma unroll
    for (int r = 0; r < 8; ++r) { c0[r] = 0.0f; c1[r] = 0.0f; }
    const float* kp0 = kb + (size_t)lo * ROWF;          // rows 0..15
    const float* kp1 = kb + (size_t)(16 + lo) * ROWF;   // rows 16..31
    #pragma unroll
    for (int kk = 0; kk < 4; ++kk) {
      // A 16x32 bf16 layout: lane<16 holds K {0..7,16..23}; lane>=16 holds K {8..15,24..31}
      const int off0 = kk * 32 + (hi ? 8 : 0);
      v16bf a0, a1;
      #pragma unroll
      for (int j = 0; j < 8; ++j) {
        a0[j]     = (__bf16)kp0[off0 + j];
        a0[8 + j] = (__bf16)kp0[off0 + 16 + j];
        a1[j]     = (__bf16)kp1[off0 + j];
        a1[8 + j] = (__bf16)kp1[off0 + 16 + j];
      }
      c0 = __builtin_amdgcn_wmma_f32_16x16x32_bf16(false, a0, false, qf[kk], (short)0, c0,
                                                   false, false);
      c1 = __builtin_amdgcn_wmma_f32_16x16x32_bf16(false, a1, false, qf[kk], (short)0, c1,
                                                   false, false);
    }

    // scale + validity mask (C layout: VGPR r -> pos r (+8 hi lanes), N=head=lane%16)
    float s0[8], s1[8];
    #pragma unroll
    for (int r = 0; r < 8; ++r) {
      int p0i = absbase + r + hi8;
      int p1i = absbase + 16 + r + hi8;
      float v0 = c0[r] * SCALE;
      float v1 = c1[r] * SCALE;
      s0[r] = (p0i < seqlen) ? v0 : -3.0e38f;
      s1[r] = (p1i < seqlen) ? v1 : -3.0e38f;
    }

    // online softmax per head: in-lane over 16 values + one xor-16 shuffle
    float mc = s0[0];
    #pragma unroll
    for (int r = 1; r < 8; ++r) mc = fmaxf(mc, s0[r]);
    #pragma unroll
    for (int r = 0; r < 8; ++r) mc = fmaxf(mc, s1[r]);
    mc = fmaxf(mc, __shfl_xor(mc, 16, 32));
    float mnew  = fmaxf(m_run, mc);
    float alpha = __expf(m_run - mnew);
    float p0[8], p1[8], ls = 0.0f;
    #pragma unroll
    for (int r = 0; r < 8; ++r) {
      p0[r] = __expf(s0[r] - mnew);
      p1[r] = __expf(s1[r] - mnew);
      ls += p0[r] + p1[r];
    }
    ls += __shfl_xor(ls, 16, 32);
    l_run = l_run * alpha + ls;
    m_run = mnew;

    // rescale O accumulators: O C-layout row r holds head (r+hi8); alpha lives in lane==head
    float frow[8];
    #pragma unroll
    for (int r = 0; r < 8; ++r) frow[r] = __shfl(alpha, r + hi8, 32);
    #pragma unroll
    for (int d = 0; d < 8; ++d) {
      #pragma unroll
      for (int r = 0; r < 8; ++r) ot[d][r] *= frow[r];
    }

    // P as A operand (M=head, K=pos 0..31): directly from the two S^T C tiles, no shuffles.
    // lane<16:  K slots 0..7 = tile0 pos 0..7,  slots 8..15 = tile1 pos 16..23
    // lane>=16: K slots 0..7 = tile0 pos 8..15, slots 8..15 = tile1 pos 24..31
    v16bf ap;
    #pragma unroll
    for (int j = 0; j < 8; ++j) { ap[j] = (__bf16)p0[j]; ap[8 + j] = (__bf16)p1[j]; }

    // O[M=head][N=d] += P * V, K=32 dense. B=V from global: element j of lane reads
    // V[pos0+j+(j&8)][dt*16 + lane%16] -> two contiguous 64B segments per instruction.
    #pragma unroll
    for (int dt = 0; dt < 8; ++dt) {
      const unsigned dcol = dt * 16 + lo;
      v16bf bvf;
      #pragma unroll
      for (int j = 0; j < 16; ++j) bvf[j] = (__bf16)v_cache[(size_t)voffs[j] + dcol];
      ot[dt] = __builtin_amdgcn_wmma_f32_16x16x32_bf16(false, ap, false, bvf, (short)0,
                                                       ot[dt], false, false);
    }
  };

  // ---- main loop: double-buffered async K staging pipelined with compute ----
  stage_chunk(0);
  for (int cc = 0; cc < 8; ++cc) {
    if (cc + 1 < 8) { stage_chunk(cc + 1); wait_stage_31(); }
    else            { wait_stage_0(); }
    process_chunk(cc);
  }

  // ---- cross-wave flash combine ----
  __syncthreads();
  if (!hi) { sm_s[w * GRP + lo] = m_run; sl_s[w * GRP + lo] = l_run; }
  {
    float* ow = o_ws + w * (GRP * DV);
    #pragma unroll
    for (int dt = 0; dt < 8; ++dt) {
      #pragma unroll
      for (int r = 0; r < 8; ++r)
        ow[(r + hi8) * DV + dt * 16 + lo] = ot[dt][r];
    }
  }
  __syncthreads();

  float* outp = out + (size_t)t * (NKV * GRP * DV) + (size_t)h * (GRP * DV);
  for (int idx = tid; idx < GRP * DV; idx += 128) {
    int head = idx >> 7;
    float M = sm_s[head];
    #pragma unroll
    for (int ww = 1; ww < 4; ++ww) M = fmaxf(M, sm_s[ww * GRP + head]);
    float Ls = 0.0f, acc = 0.0f;
    #pragma unroll
    for (int ww = 0; ww < 4; ++ww) {
      float f = __expf(sm_s[ww * GRP + head] - M);
      Ls  += sl_s[ww * GRP + head] * f;
      acc += o_ws[ww * (GRP * DV) + idx] * f;
    }
    outp[idx] = acc / Ls;
  }
}

extern "C" void kernel_launch(void* const* d_in, const int* in_sizes, int n_in,
                              void* d_out, int out_size, void* d_ws, size_t ws_size,
                              hipStream_t stream) {
  const float* q    = (const float*)d_in[0];
  const int*   sel  = (const int*)d_in[1];
  const float* kc   = (const float*)d_in[2];
  const float* vc   = (const float*)d_in[3];
  const int*   rtt  = (const int*)d_in[4];
  const int*   slen = (const int*)d_in[5];
  float* out = (float*)d_out;
  (void)in_sizes; (void)n_in; (void)out_size; (void)d_ws; (void)ws_size;

  dim3 grid(TQ, NKV);
  dim3 block(128);
  SelAttn_gfx1250_kernel<<<grid, block, 0, stream>>>(q, sel, kc, vc, rtt, slen, out);
}
